// PoolingConcatFusion_87393994539108
// MI455X (gfx1250) — matrix operand order repaired and checked by reading, not compiled, not run on hardware
//
#include <hip/hip_runtime.h>
#include <hip/hip_bf16.h>

typedef __attribute__((ext_vector_type(16))) _Float16 v16h;
typedef __attribute__((ext_vector_type(8)))  _Float16 v8h;
typedef __attribute__((ext_vector_type(8)))  float    v8f;

#define NUM_GRAPHS 512
#define DIM 128
#define LDSS 264          // LDS row stride in halves (16B-aligned, avoids bank conflicts)
#define NEG_SLOPE 0.01f

// ---------------------------------------------------------------------------
// Kernel 1: per-graph pooling. batch is sorted, so block g binary-searches its
// [start,end) range and thread t reduces feature column t (coalesced reads).
// P layout per graph: [t_max(128) | t_mean(128) | g_max(128) | g_mean(128)]
// ---------------------------------------------------------------------------
__device__ __forceinline__ int lower_bound_i32(const int* __restrict__ a, int n, int val) {
    int lo = 0, hi = n;
    while (lo < hi) { int mid = (lo + hi) >> 1; if (a[mid] < val) lo = mid + 1; else hi = mid; }
    return lo;
}

__global__ __launch_bounds__(128) void pool_kernel(
    const float* __restrict__ ht, const float* __restrict__ hg,
    const int* __restrict__ batch, int N, float* __restrict__ P)
{
    const int g = blockIdx.x;
    const int t = threadIdx.x;                  // feature 0..127
    const int s = lower_bound_i32(batch, N, g);
    const int e = lower_bound_i32(batch, N, g + 1);

    float tmax = -__builtin_inff(), gmax = -__builtin_inff();
    float tsum = 0.f, gsum = 0.f;
    for (int n = s; n < e; ++n) {
        float a = ht[(long)n * DIM + t];
        float b = hg[(long)n * DIM + t];
        tmax = fmaxf(tmax, a); tsum += a;
        gmax = fmaxf(gmax, b); gsum += b;
    }
    float cnt = (float)((e - s) > 1 ? (e - s) : 1);
    P[g * 512 + t]         = tmax;
    P[g * 512 + 128 + t]   = tsum / cnt;
    P[g * 512 + 256 + t]   = gmax;
    P[g * 512 + 384 + t]   = gsum / cnt;
}

// ---------------------------------------------------------------------------
// Kernel 2: weight prep.
//  W1t f16 [256 n][256 k] col-major: node-feature rows of W1 = rows {0..127, 384..511}
//  W1p f32 [512 j][256 c]: pooled-feature rows of W1 = rows {128..383, 512..767}
//  W2t f16 [128 n][256 k] col-major
// ---------------------------------------------------------------------------
__global__ __launch_bounds__(256) void prep_weights(
    const float* __restrict__ W1, const float* __restrict__ W2,
    _Float16* __restrict__ W1t, float* __restrict__ W1p, _Float16* __restrict__ W2t)
{
    int idx = blockIdx.x * 256 + threadIdx.x;
    if (idx < 256 * 256) {                        // W1t
        int n = idx >> 8, k = idx & 255;
        int row = (k < 128) ? k : (256 + k);      // 384 + (k-128)
        W1t[n * 256 + k] = (_Float16)W1[row * 256 + n];
    } else if (idx < 256 * 256 + 512 * 256) {     // W1p
        int i = idx - 256 * 256;
        int j = i >> 8, c = i & 255;
        int row = (j < 256) ? (128 + j) : (256 + j);   // covers 128..383, 512..767
        W1p[j * 256 + c] = W1[row * 256 + c];
    } else if (idx < 256 * 256 + 512 * 256 + 128 * 256) {  // W2t
        int i = idx - (256 * 256 + 512 * 256);
        int n = i >> 8, k = i & 255;
        W2t[n * 256 + k] = (_Float16)W2[k * DIM + n];
    }
}

// ---------------------------------------------------------------------------
// Kernel 3: gbias[g][c] = b1[c] + sum_j P[g][j] * W1p[j][c]   (512x512x256 fp32)
// ---------------------------------------------------------------------------
__global__ __launch_bounds__(256) void gbias_kernel(
    const float* __restrict__ P, const float* __restrict__ W1p,
    const float* __restrict__ b1, float* __restrict__ gbias)
{
    __shared__ float prow[512];
    const int g = blockIdx.x, c = threadIdx.x;
    for (int j = c; j < 512; j += 256) prow[j] = P[g * 512 + j];
    __syncthreads();
    float acc = b1[c];
#pragma unroll 8
    for (int j = 0; j < 512; ++j) acc += prow[j] * W1p[j * 256 + c];
    gbias[g * 256 + c] = acc;
}

// ---------------------------------------------------------------------------
// Kernel 4: fused per-node MLP via WMMA.
// 256 threads = 8 waves, 32 nodes (2 M-tiles) per block.
// Layer 1: wave w -> M-tiles {0,1} x N-tiles {2w, 2w+1}: 2x2 register tiling,
//          each A/B fragment feeds 2 WMMAs (4 WMMA per 4 fragment loads).
// Layer 2: wave w -> M-tiles {0,1} x N-tile w: B fragment reused twice.
// ---------------------------------------------------------------------------
__device__ __forceinline__ v16h load_a_frag(const _Float16* __restrict__ row, int k0, int hl) {
    // a[j] = row[k0 + 16*hl + j]  (contiguous k order, matches B)
    const _Float16* p = row + k0 + 16 * hl;
    v8h lo = *(const v8h*)p;
    v8h hi = *(const v8h*)(p + 8);
    v16h a;
#pragma unroll
    for (int j = 0; j < 8; ++j) { a[j] = lo[j]; a[j + 8] = hi[j]; }
    return a;
}

#define WMMA_F16(A, B, C) \
    __builtin_amdgcn_wmma_f32_16x16x32_f16(false, (A), false, (B), (short)0, (C), false, false)

__global__ __launch_bounds__(256) void fused_mlp(
    const float* __restrict__ h_topo, const float* __restrict__ h_geom,
    const int* __restrict__ batch,
    const _Float16* __restrict__ W1t, const _Float16* __restrict__ W2t,
    const float* __restrict__ gbias, const float* __restrict__ b2,
    float* __restrict__ out, int N)
{
    __shared__ __align__(32) _Float16 Alds[32 * LDSS];
    __shared__ __align__(32) _Float16 Hlds[32 * LDSS];
    __shared__ int   gids[32];
    __shared__ float b2s[DIM];

    const int tid   = threadIdx.x;
    const int node0 = blockIdx.x * 32;

    if (tid < 32) {
        int n = node0 + tid; if (n >= N) n = N - 1;
        gids[tid] = batch[n];
    }
    if (tid < DIM) b2s[tid] = b2[tid];

    // Stage A = [h_topo | h_geom] (32 x 256) as f16 in LDS.
    for (int i = tid; i < 32 * 256; i += 256) {
        int r = i >> 8, c = i & 255;
        int n = node0 + r; if (n >= N) n = N - 1;
        float v = (c < DIM) ? h_topo[(long)n * DIM + c] : h_geom[(long)n * DIM + (c - DIM)];
        Alds[r * LDSS + c] = (_Float16)v;
    }
    __syncthreads();

    const int lane = tid & 31;
    const int wave = tid >> 5;
    const int hl   = lane >> 4;   // lane half (0/1)
    const int mn   = lane & 15;   // row (for A/C) or col (for B/C)

    // ---- Layer 1: 2 M-tiles x 2 N-tiles per wave
    v8f acc00 = {}, acc01 = {}, acc10 = {}, acc11 = {};
    const int nc0 = 32 * wave;
#pragma unroll
    for (int k0 = 0; k0 < 256; k0 += 32) {
        v16h a0 = load_a_frag(&Alds[mn * LDSS],        k0, hl);
        v16h a1 = load_a_frag(&Alds[(16 + mn) * LDSS], k0, hl);
        v16h b0 = *(const v16h*)&W1t[(nc0 + mn)      * 256 + k0 + 16 * hl];
        v16h b1 = *(const v16h*)&W1t[(nc0 + 16 + mn) * 256 + k0 + 16 * hl];
        acc00 = WMMA_F16(a0, b0, acc00);
        acc01 = WMMA_F16(a0, b1, acc01);
        acc10 = WMMA_F16(a1, b0, acc10);
        acc11 = WMMA_F16(a1, b1, acc11);
    }
    // Epilogue: + gbias[graph] (b1 folded in), LeakyReLU, f16 -> LDS
#pragma unroll
    for (int r = 0; r < 8; ++r) {
        int m  = r + 8 * hl;
        int g0 = gids[m];
        int g1 = gids[16 + m];
        float v;
        v = acc00[r] + gbias[g0 * 256 + nc0 + mn];
        v = (v >= 0.f) ? v : NEG_SLOPE * v;
        Hlds[m * LDSS + nc0 + mn] = (_Float16)v;
        v = acc01[r] + gbias[g0 * 256 + nc0 + 16 + mn];
        v = (v >= 0.f) ? v : NEG_SLOPE * v;
        Hlds[m * LDSS + nc0 + 16 + mn] = (_Float16)v;
        v = acc10[r] + gbias[g1 * 256 + nc0 + mn];
        v = (v >= 0.f) ? v : NEG_SLOPE * v;
        Hlds[(16 + m) * LDSS + nc0 + mn] = (_Float16)v;
        v = acc11[r] + gbias[g1 * 256 + nc0 + 16 + mn];
        v = (v >= 0.f) ? v : NEG_SLOPE * v;
        Hlds[(16 + m) * LDSS + nc0 + 16 + mn] = (_Float16)v;
    }
    __syncthreads();

    // ---- Layer 2: 2 M-tiles x 1 N-tile per wave (B fragment reused)
    v8f c0 = {}, c1 = {};
    const int oc0 = 16 * wave;
#pragma unroll
    for (int k0 = 0; k0 < 256; k0 += 32) {
        v16h a0 = load_a_frag(&Hlds[mn * LDSS],        k0, hl);
        v16h a1 = load_a_frag(&Hlds[(16 + mn) * LDSS], k0, hl);
        v16h b  = *(const v16h*)&W2t[(oc0 + mn) * 256 + k0 + 16 * hl];
        c0 = WMMA_F16(a0, b, c0);
        c1 = WMMA_F16(a1, b, c1);
    }
#pragma unroll
    for (int r = 0; r < 8; ++r) {
        int m = r + 8 * hl;
        float bias = b2s[oc0 + mn];
        int n0 = node0 + m;
        if (n0 < N)
            out[(long)n0 * DIM + oc0 + mn] = c0[r] + bias;
        int n1 = node0 + 16 + m;
        if (n1 < N)
            out[(long)n1 * DIM + oc0 + mn] = c1[r] + bias;
    }
}

// ---------------------------------------------------------------------------
extern "C" void kernel_launch(void* const* d_in, const int* in_sizes, int n_in,
                              void* d_out, int out_size, void* d_ws, size_t ws_size,
                              hipStream_t stream) {
    const float* h_topo = (const float*)d_in[0];
    const float* h_geom = (const float*)d_in[1];
    const int*   batch  = (const int*)d_in[2];
    const float* W1     = (const float*)d_in[3];
    const float* b1     = (const float*)d_in[4];
    const float* W2     = (const float*)d_in[5];
    const float* b2     = (const float*)d_in[6];
    float*       out    = (float*)d_out;

    const int N = in_sizes[0] / DIM;

    char* ws = (char*)d_ws;
    float*    P     = (float*)(ws);                          // 512*512*4   = 1048576
    float*    W1p   = (float*)(ws + 1048576);                // 512*256*4   =  524288
    float*    gbias = (float*)(ws + 1048576 + 524288);       // 512*256*4   =  524288
    _Float16* W1t   = (_Float16*)(ws + 2097152);             // 256*256*2   =  131072
    _Float16* W2t   = (_Float16*)(ws + 2097152 + 131072);    // 128*256*2   =   65536

    pool_kernel<<<dim3(NUM_GRAPHS), dim3(128), 0, stream>>>(h_topo, h_geom, batch, N, P);
    prep_weights<<<dim3((256*256 + 512*256 + 128*256) / 256), dim3(256), 0, stream>>>(W1, W2, W1t, W1p, W2t);
    gbias_kernel<<<dim3(NUM_GRAPHS), dim3(256), 0, stream>>>(P, W1p, b1, gbias);

    const int blocks = (N + 31) / 32;
    fused_mlp<<<dim3(blocks), dim3(256), 0, stream>>>(h_topo, h_geom, batch, W1t, W2t, gbias, b2, out, N);
}
